// matchingNet_9723805958606
// MI455X (gfx1250) — matrix operand order repaired and checked by reading, not compiled, not run on hardware
//
#include <hip/hip_runtime.h>
#include <hip/hip_bf16.h>
#include <math.h>

typedef __attribute__((ext_vector_type(16))) _Float16 v16h;
typedef __attribute__((ext_vector_type(8)))  float    v8f;
typedef _Float16 h16;

#define NB 1024
#define HD 64
#define WAY 20

__device__ __forceinline__ float sigm(float x){ return 1.0f/(1.0f+__expf(-x)); }

// Generic pointer -> 32-bit LDS byte offset (addrspace(3) pointers are LDS offsets).
__device__ __forceinline__ unsigned lds_off32(const void* p){
    return (unsigned)(unsigned long long)(__attribute__((address_space(3))) const void*)p;
}

// gfx1250 async global->LDS copy of 16 bytes, tracked by ASYNCcnt (ISA 15.18.3 op 98).
__device__ __forceinline__ void async_copy_b128(const void* gptr, const void* lptr){
    unsigned       loff  = lds_off32(lptr);
    unsigned long long ga = (unsigned long long)gptr;
    asm volatile("global_load_async_to_lds_b128 %0, %1, off"
                 :: "v"(loff), "v"(ga) : "memory");
}
__device__ __forceinline__ void async_wait0(){
    asm volatile("s_wait_asynccnt 0x0" ::: "memory");
}

// ISA 7.12.2: 16-bit A-matrix 16x32 per-lane K map. lane half t (0..15) -> K.
__device__ __forceinline__ int a_kmap(int lane, int t){
    int grp = (lane >> 4) & 1;       // lanes 0-15 vs 16-31
    int v   = t >> 1;                // vgpr index 0..7
    int par = t & 1;
    int base = (v < 4) ? 0 : 16;
    return base + ((v & 3) << 1) + par + (grp << 3);
}

// ---------------- conv1 (1->64, 28->26, BN stats over batch, recompute to avoid big buffer)
__global__ void k_conv1_stats(const float* __restrict__ im, const float* __restrict__ w,
                              const float* __restrict__ bias, const float* __restrict__ g,
                              const float* __restrict__ beta, float* __restrict__ sc,
                              float* __restrict__ sh){
    int c = blockIdx.x, tid = threadIdx.x;
    float wr[9];
    #pragma unroll
    for (int i = 0; i < 9; i++) wr[i] = w[c*9 + i];
    float bc = bias[c];
    const int total = NB * 26 * 26;
    float s = 0.f, ss = 0.f;
    for (int i = tid; i < total; i += 256){
        int b = i / 676; int p = i - b*676; int oy = p / 26; int ox = p - oy*26;
        const float* ip = im + b*784 + oy*28 + ox;
        float y = bc;
        #pragma unroll
        for (int kh = 0; kh < 3; kh++)
            #pragma unroll
            for (int kw = 0; kw < 3; kw++) y += ip[kh*28 + kw] * wr[kh*3 + kw];
        s += y; ss += y*y;
    }
    __shared__ float r0[256], r1[256];
    r0[tid] = s; r1[tid] = ss; __syncthreads();
    for (int d = 128; d > 0; d >>= 1){
        if (tid < d){ r0[tid] += r0[tid+d]; r1[tid] += r1[tid+d]; }
        __syncthreads();
    }
    if (tid == 0){
        float mean = r0[0] / (float)total;
        float var  = r1[0] / (float)total - mean*mean;
        float rstd = rsqrtf(var + 1e-5f);
        float scl  = g[c] * rstd;
        sc[c] = scl; sh[c] = beta[c] - mean*scl;
    }
}

__global__ void k_conv1_bnpool(const float* __restrict__ im, const float* __restrict__ w,
                               const float* __restrict__ bias, const float* __restrict__ sc,
                               const float* __restrict__ sh, h16* __restrict__ out){
    int idx = blockIdx.x*256 + threadIdx.x;
    if (idx >= NB*64*169) return;
    int pw = idx % 13; int t = idx / 13; int ph = t % 13; t /= 13; int c = t % 64; int b = t / 64;
    float wr[9];
    #pragma unroll
    for (int i = 0; i < 9; i++) wr[i] = w[c*9 + i];
    float bc = bias[c], scl = sc[c], shf = sh[c];
    float m = -1e30f;
    #pragma unroll
    for (int dy = 0; dy < 2; dy++){
        #pragma unroll
        for (int dx = 0; dx < 2; dx++){
            int oy = ph*2 + dy, ox = pw*2 + dx;
            const float* ip = im + b*784 + oy*28 + ox;
            float y = bc;
            #pragma unroll
            for (int kh = 0; kh < 3; kh++)
                #pragma unroll
                for (int kw = 0; kw < 3; kw++) y += ip[kh*28 + kw] * wr[kh*3 + kw];
            m = fmaxf(m, y*scl + shf);
        }
    }
    out[idx] = (h16)fmaxf(m, 0.f);
}

// ---------------- weight packing into per-lane WMMA A-fragments: [kstep18][mtile4][lane32][t16]
__global__ void k_pack_wfrag(const float* __restrict__ w, h16* __restrict__ frag){
    int idx = blockIdx.x*256 + threadIdx.x;
    if (idx >= 18*4*32*16) return;
    int t = idx & 15; int lane = (idx >> 4) & 31; int mt = (idx >> 9) & 3; int ks = idx >> 11;
    int m = mt*16 + (lane & 15);
    int k = ks*32 + ((lane >> 4) << 4) + t;
    int ic = k / 9; int r = k - ic*9; int kh = r / 3; int kw = r - kh*3;
    frag[idx] = (h16)w[(m*64 + ic)*9 + kh*3 + kw];
}

// ---------------- generic 64->64 k3 conv as implicit GEMM with WMMA (M=64, K=576, N=OH*OW)
__global__ void __launch_bounds__(256) k_conv_wmma(const h16* __restrict__ xin,
                                                   const h16* __restrict__ wfrag,
                                                   const float* __restrict__ bias,
                                                   float* __restrict__ yout,
                                                   int IH, int IW, int OH, int OW){
    __shared__ __align__(16) h16 xs[64*13*13];
    int b = blockIdx.x, tid = threadIdx.x;
    int IHW = IH*IW, NP = OH*OW, NT = (NP + 15) >> 4;
    int total = 64*IHW;                                  // multiple of 8 halves
    const h16* gbase = xin + (size_t)b*total;
    for (int i = tid*8; i < total; i += 256*8)           // async tile stage, 16B per lane-op
        async_copy_b128(gbase + i, xs + i);
    async_wait0();
    __syncthreads();
    int lane = tid & 31, wave = tid >> 5;
    int ln16 = lane & 15, grp = lane >> 4;
    for (int tile = wave; tile < 4*NT; tile += 8){
        int mt = tile & 3, nt = tile >> 2;
        int n = nt*16 + ln16;
        bool valid = n < NP;
        int oy = valid ? n / OW : 0;
        int ox = valid ? n - oy*OW : 0;
        v8f c = {};
        for (int ks = 0; ks < 18; ks++){
            v16h a = *(const v16h*)(wfrag + ((size_t)(ks*4 + mt)*32 + lane)*16);
            v16h bb{};
            int kb = ks*32 + (grp << 4);
            #pragma unroll
            for (int t = 0; t < 16; t++){
                int k = kb + t;
                int ic = k / 9; int r = k - ic*9; int kh = r / 3; int kw = r - kh*3;
                bb[t] = valid ? xs[ic*IHW + (oy + kh)*IW + (ox + kw)] : (h16)0.0f;
            }
            c = __builtin_amdgcn_wmma_f32_16x16x32_f16(false, a, false, bb, (short)0, c, false, false);
        }
        if (valid){
            #pragma unroll
            for (int r = 0; r < 8; r++){
                int oc = mt*16 + r + (grp << 3);
                yout[((size_t)b*64 + oc)*NP + n] = c[r] + bias[oc];
            }
        }
    }
}

// ---------------- per-channel BN stats over stored conv output
__global__ void k_stats(const float* __restrict__ y, int NP, const float* __restrict__ g,
                        const float* __restrict__ beta, float* __restrict__ sc,
                        float* __restrict__ sh){
    int c = blockIdx.x, tid = threadIdx.x;
    int total = NB * NP;
    float s = 0.f, ss = 0.f;
    for (int i = tid; i < total; i += 256){
        int b = i / NP; int p = i - b*NP;
        float v = y[((size_t)b*64 + c)*NP + p];
        s += v; ss += v*v;
    }
    __shared__ float r0[256], r1[256];
    r0[tid] = s; r1[tid] = ss; __syncthreads();
    for (int d = 128; d > 0; d >>= 1){
        if (tid < d){ r0[tid] += r0[tid+d]; r1[tid] += r1[tid+d]; }
        __syncthreads();
    }
    if (tid == 0){
        float mean = r0[0] / (float)total;
        float var  = r1[0] / (float)total - mean*mean;
        float rstd = rsqrtf(var + 1e-5f);
        float scl  = g[c] * rstd;
        sc[c] = scl; sh[c] = beta[c] - mean*scl;
    }
}

// ---------------- BN apply + ReLU + 2x2 maxpool; writes f16 (next conv) or f32 (final features)
__global__ void k_bnpool(const float* __restrict__ y, const float* __restrict__ sc,
                         const float* __restrict__ sh, h16* __restrict__ out16,
                         float* __restrict__ out32, int OH, int OW, int PH, int PW){
    int idx = blockIdx.x*256 + threadIdx.x;
    int total = NB*64*PH*PW;
    if (idx >= total) return;
    int pw = idx % PW; int t = idx / PW; int ph = t % PH; t /= PH; int c = t % 64; int b = t / 64;
    const float* yp = y + ((size_t)b*64 + c)*OH*OW;
    float scl = sc[c], shf = sh[c];
    float m = -1e30f;
    #pragma unroll
    for (int dy = 0; dy < 2; dy++)
        #pragma unroll
        for (int dx = 0; dx < 2; dx++)
            m = fmaxf(m, yp[(ph*2 + dy)*OW + (pw*2 + dx)] * scl + shf);
    m = fmaxf(m, 0.f);
    if (out32) out32[idx] = m;
    else       out16[idx] = (h16)m;
}

// ---------------- batched gate pre-compute: out[n][g] = bias(+bias2) + sum_h x[n][h]*W(g,h)
__global__ void k_gates(const float* __restrict__ x, const float* __restrict__ W,
                        const float* __restrict__ b1, const float* __restrict__ b2,
                        float* __restrict__ out, int transposed){
    __shared__ float xr[64];
    int n = blockIdx.x, g = threadIdx.x;
    if (g < 64) xr[g] = x[n*64 + g];
    __syncthreads();
    float acc = b1[g] + (b2 ? b2[g] : 0.f);
    if (transposed){                       // W is [64][256]
        for (int h = 0; h < 64; h++) acc += xr[h] * W[h*256 + g];
    } else {                               // W is [256][64]
        const float* wr = W + g*64;
        for (int h = 0; h < 64; h++) acc += xr[h] * wr[h];
    }
    out[n*256 + g] = acc;
}

// ---------------- sequential LSTM direction: 1 workgroup, Wh resident in LDS
__global__ void k_lstm(const float* __restrict__ xw, const float* __restrict__ wh,
                       const float* __restrict__ h0, const float* __restrict__ c0,
                       float* __restrict__ hout, int reverse){
    extern __shared__ char smem[];
    float* whs = (float*)smem;          // 256*64
    float* hs  = whs + 16384;           // 64
    float* cs  = hs + 64;               // 64
    float* gs  = cs + 64;               // 256
    int tid = threadIdx.x;
    for (int i = tid; i < 16384; i += 256) whs[i] = wh[i];
    if (tid < 64){ hs[tid] = h0[tid]; cs[tid] = c0[tid]; }
    __syncthreads();
    for (int t = 0; t < NB; t++){
        int tin = reverse ? (NB - 1 - t) : t;
        float acc = xw[tin*256 + tid];
        const float* wr = whs + tid*64;
        for (int h = 0; h < 64; h++) acc += wr[h] * hs[h];
        gs[tid] = acc;
        __syncthreads();
        if (tid < 64){
            float i_ = sigm(gs[tid]);
            float f_ = sigm(gs[64 + tid]);
            float g_ = tanhf(gs[128 + tid]);
            float o_ = sigm(gs[192 + tid]);
            float cn = f_*cs[tid] + i_*g_; cs[tid] = cn;
            float hn = o_*tanhf(cn); hs[tid] = hn;
            hout[t*64 + tid] = hn;
        }
        __syncthreads();
    }
}

// ---------------- cond_train = hf + hb + ftr, also f16 copy + row norm
__global__ void k_combine(const float* __restrict__ hf, const float* __restrict__ hb,
                          const float* __restrict__ ftr, h16* __restrict__ cond16,
                          float* __restrict__ norms){
    int i = blockIdx.x, t = threadIdx.x;
    float v = hf[i*64 + t] + hb[i*64 + t] + ftr[i*64 + t];
    cond16[i*64 + t] = (h16)v;
    __shared__ float red[64];
    red[t] = v*v; __syncthreads();
    for (int d = 32; d > 0; d >>= 1){ if (t < d) red[t] += red[t+d]; __syncthreads(); }
    if (t == 0) norms[i] = sqrtf(red[0]);
}

__global__ void k_norms16(const h16* __restrict__ x, float* __restrict__ norms){
    int i = blockIdx.x, t = threadIdx.x;
    float v = (float)x[i*64 + t];
    __shared__ float red[64];
    red[t] = v*v; __syncthreads();
    for (int d = 32; d > 0; d >>= 1){ if (t < d) red[t] += red[t+d]; __syncthreads(); }
    if (t == 0) norms[i] = sqrtf(red[0]);
}

// ---------------- attLSTM: single 1024-thread workgroup; support set + WhhT resident in LDS
__global__ void __launch_bounds__(1024) k_attlstm(const h16* __restrict__ cond16g,
                                                  const float* __restrict__ xa,
                                                  const float* __restrict__ fte,
                                                  const float* __restrict__ whh,
                                                  const float* __restrict__ h0,
                                                  const float* __restrict__ c0,
                                                  const float* __restrict__ r0,
                                                  h16* __restrict__ ct16){
    extern __shared__ char smem[];
    h16*   cs16 = (h16*)smem;                          // 1024*64 f16 = 128KB
    float* whhT = (float*)(smem + 131072);             // 256*128 f32 = 128KB (transposed)
    float* att  = (float*)(smem + 262144);             // 1024
    float* part = att + 1024;                          // 1024
    float* hsh  = part + 1024;                         // 64
    float* csh  = hsh + 64;
    float* rsh  = csh + 64;
    float* xsh  = rsh + 64;
    float* gsh  = xsh + 64;                            // 256
    float* wred = gsh + 256;                           // 34
    int tid = threadIdx.x, lane = tid & 31, wv = tid >> 5;
    for (int i = tid*8; i < 1024*64; i += 1024*8)      // async stage of support set
        async_copy_b128(cond16g + i, cs16 + i);
    for (int i = tid; i < 32768; i += 1024){ int g = i >> 7, j = i & 127; whhT[i] = whh[j*256 + g]; }
    if (tid < 64){ hsh[tid] = h0[tid]; csh[tid] = c0[tid]; rsh[tid] = r0[tid]; }
    async_wait0();
    __syncthreads();
    for (int t = 0; t < NB; t++){
        if (tid < 64) xsh[tid] = fte[t*64 + tid];
        __syncthreads();
        if (tid < 256){
            float acc = xa[t*256 + tid];
            const float* wr = whhT + tid*128;
            for (int j = 0; j < 64; j++) acc += wr[j]      * hsh[j];
            for (int j = 0; j < 64; j++) acc += wr[64 + j] * rsh[j];
            gsh[tid] = acc;
        }
        __syncthreads();
        if (tid < 64){                                   // cell gate order f,i,o,g
            float f_ = sigm(gsh[tid]);
            float i_ = sigm(gsh[64 + tid]);
            float o_ = sigm(gsh[128 + tid]);
            float g_ = tanhf(gsh[192 + tid]);
            float cn = f_*csh[tid] + i_*g_; csh[tid] = cn;
            float hn = o_*tanhf(cn) + xsh[tid];          // residual add of testF
            hsh[tid] = hn;
            ct16[t*64 + tid] = (h16)hn;
        }
        __syncthreads();
        // attention logits over the 1024-row support set (all in LDS)
        float s = 0.f;
        const h16* cr = cs16 + tid*64;
        for (int k = 0; k < 64; k++) s += (float)cr[k] * hsh[k];
        float m = s;
        for (int d = 16; d > 0; d >>= 1) m = fmaxf(m, __shfl_xor(m, d));
        if (lane == 0) wred[wv] = m;
        __syncthreads();
        if (tid < 32){
            float mm = wred[tid];
            for (int d = 16; d > 0; d >>= 1) mm = fmaxf(mm, __shfl_xor(mm, d));
            if (tid == 0) wred[32] = mm;
        }
        __syncthreads();
        float e = __expf(s - wred[32]);
        att[tid] = e;
        float se = e;
        for (int d = 16; d > 0; d >>= 1) se += __shfl_xor(se, d);
        if (lane == 0) wred[wv] = se;
        __syncthreads();
        if (tid < 32){
            float s2 = wred[tid];
            for (int d = 16; d > 0; d >>= 1) s2 += __shfl_xor(s2, d);
            if (tid == 0) wred[32] = s2;
        }
        __syncthreads();
        float ssum = wred[32];
        int chunk = tid >> 6, k = tid & 63, base = chunk << 6;
        float acc = 0.f;
        for (int i = 0; i < 64; i++) acc += att[base + i] * (float)cs16[(base + i)*64 + k];
        part[(chunk << 6) + k] = acc;
        __syncthreads();
        if (tid < 64){
            float rv = 0.f;
            #pragma unroll
            for (int q = 0; q < 16; q++) rv += part[(q << 6) + tid];
            rsh[tid] = rv / ssum;
        }
        __syncthreads();
    }
}

// ---------------- cosine similarity GEMM (1024x1024x64) via WMMA
__global__ void __launch_bounds__(128) k_sim(const h16* __restrict__ ct, const h16* __restrict__ cn,
                                             const float* __restrict__ nt, const float* __restrict__ nn,
                                             float* __restrict__ sim){
    int tid = threadIdx.x, lane = tid & 31, wave = tid >> 5;
    int mt = blockIdx.x;
    int ntile = blockIdx.y*4 + wave;
    int ln16 = lane & 15, grp = lane >> 4;
    v8f c = {};
    #pragma unroll
    for (int ks = 0; ks < 2; ks++){
        v16h a{};
        int arow = (mt*16 + ln16)*64;
        #pragma unroll
        for (int t = 0; t < 16; t++) a[t] = ct[arow + ks*32 + (grp << 4) + t];
        // B fragment: k = ks*32 + grp*16 + t for t=0..15 -> 16 contiguous, 32B-aligned halves
        v16h b = *(const v16h*)(cn + (ntile*16 + ln16)*64 + ks*32 + (grp << 4));
        c = __builtin_amdgcn_wmma_f32_16x16x32_f16(false, a, false, b, (short)0, c, false, false);
    }
    #pragma unroll
    for (int r = 0; r < 8; r++){
        int m = mt*16 + r + (grp << 3);
        int n = ntile*16 + ln16;
        float den = nt[m] * nn[n];
        sim[(size_t)m*1024 + n] = c[r] / fmaxf(den, 1e-8f);
    }
}

// ---------------- per-row softmax over support + label mix
__global__ void k_softmax_label(const float* __restrict__ sim, const float* __restrict__ lab,
                                float* __restrict__ out){
    int t = blockIdx.x, tid = threadIdx.x;
    __shared__ float red[256];
    __shared__ float bm, bs;
    const float* sr = sim + (size_t)t*1024;
    float m = -1e30f;
    for (int i = tid; i < 1024; i += 256) m = fmaxf(m, sr[i]);
    red[tid] = m; __syncthreads();
    for (int d = 128; d > 0; d >>= 1){ if (tid < d) red[tid] = fmaxf(red[tid], red[tid+d]); __syncthreads(); }
    if (tid == 0) bm = red[0];
    __syncthreads();
    float lsum = 0.f, lo[WAY];
    #pragma unroll
    for (int w = 0; w < WAY; w++) lo[w] = 0.f;
    for (int i = tid; i < 1024; i += 256){
        float e = __expf(sr[i] - bm);
        lsum += e;
        #pragma unroll
        for (int w = 0; w < WAY; w++) lo[w] += e * lab[i*WAY + w];
    }
    red[tid] = lsum; __syncthreads();
    for (int d = 128; d > 0; d >>= 1){ if (tid < d) red[tid] += red[tid+d]; __syncthreads(); }
    if (tid == 0) bs = red[0];
    __syncthreads();
    for (int w = 0; w < WAY; w++){
        red[tid] = lo[w]; __syncthreads();
        for (int d = 128; d > 0; d >>= 1){ if (tid < d) red[tid] += red[tid+d]; __syncthreads(); }
        if (tid == 0) out[t*WAY + w] = red[0] / bs;
        __syncthreads();
    }
}

extern "C" void kernel_launch(void* const* d_in, const int* in_sizes, int n_in,
                              void* d_out, int out_size, void* d_ws, size_t ws_size,
                              hipStream_t stream) {
    (void)in_sizes; (void)n_in; (void)out_size; (void)ws_size;
    const float* trainIm    = (const float*)d_in[0];
    const float* trainLabel = (const float*)d_in[1];
    const float* testIm     = (const float*)d_in[2];
    const float* c1w = (const float*)d_in[4];  const float* c1b = (const float*)d_in[5];
    const float* bn1g= (const float*)d_in[6];  const float* bn1b= (const float*)d_in[7];
    const float* c2w = (const float*)d_in[8];  const float* c2b = (const float*)d_in[9];
    const float* bn2g= (const float*)d_in[10]; const float* bn2b= (const float*)d_in[11];
    const float* c3w = (const float*)d_in[12]; const float* c3b = (const float*)d_in[13];
    const float* bn3g= (const float*)d_in[14]; const float* bn3b= (const float*)d_in[15];
    const float* wi_f= (const float*)d_in[16]; const float* wh_f= (const float*)d_in[17];
    const float* bi_f= (const float*)d_in[18]; const float* bh_f= (const float*)d_in[19];
    const float* wi_b= (const float*)d_in[20]; const float* wh_b= (const float*)d_in[21];
    const float* bi_b= (const float*)d_in[22]; const float* bh_b= (const float*)d_in[23];
    const float* h0  = (const float*)d_in[24]; const float* c0  = (const float*)d_in[25];
    const float* att_wih = (const float*)d_in[26];
    const float* att_whh = (const float*)d_in[27];
    const float* att_b   = (const float*)d_in[28];
    const float* att_h0  = (const float*)d_in[29];
    const float* att_c0  = (const float*)d_in[30];
    const float* att_r0  = (const float*)d_in[31];
    float* out = (float*)d_out;

    char* ws = (char*)d_ws;
    size_t off = 0;
    auto take = [&](size_t bytes) -> char* {
        char* p = ws + off;
        off = (off + bytes + 255) & ~(size_t)255;
        return p;
    };
    h16*   x2     = (h16*)  take((size_t)NB*64*169*2);
    float* y2     = (float*)take((size_t)NB*64*121*4);
    h16*   x3     = (h16*)  take((size_t)NB*64*25*2);
    float* y3     = (float*)take((size_t)NB*64*9*4);
    float* ftr    = (float*)take((size_t)NB*64*4);
    float* fte    = (float*)take((size_t)NB*64*4);
    h16*   w2frag = (h16*)  take((size_t)18*4*32*16*2);
    h16*   w3frag = (h16*)  take((size_t)18*4*32*16*2);
    float* bn_sc  = (float*)take(64*4);
    float* bn_sh  = (float*)take(64*4);
    float* xw_f   = (float*)take((size_t)NB*256*4);
    float* xw_b   = (float*)take((size_t)NB*256*4);
    float* xa     = (float*)take((size_t)NB*256*4);
    float* hf     = (float*)take((size_t)NB*64*4);
    float* hb     = (float*)take((size_t)NB*64*4);
    h16*   ctr16  = (h16*)  take((size_t)NB*64*2);
    h16*   cte16  = (h16*)  take((size_t)NB*64*2);
    float* ntr    = (float*)take((size_t)NB*4);
    float* nte    = (float*)take((size_t)NB*4);
    float* sim    = (float*)take((size_t)NB*NB*4);

    k_pack_wfrag<<<144, 256, 0, stream>>>(c2w, w2frag);
    k_pack_wfrag<<<144, 256, 0, stream>>>(c3w, w3frag);

    const float* ims[2]   = { trainIm, testIm };
    float*       feats[2] = { ftr, fte };
    for (int s = 0; s < 2; s++){
        k_conv1_stats <<<64, 256, 0, stream>>>(ims[s], c1w, c1b, bn1g, bn1b, bn_sc, bn_sh);
        k_conv1_bnpool<<<(NB*64*169 + 255)/256, 256, 0, stream>>>(ims[s], c1w, c1b, bn_sc, bn_sh, x2);
        k_conv_wmma   <<<NB, 256, 0, stream>>>(x2, w2frag, c2b, y2, 13, 13, 11, 11);
        k_stats       <<<64, 256, 0, stream>>>(y2, 121, bn2g, bn2b, bn_sc, bn_sh);
        k_bnpool      <<<(NB*64*25 + 255)/256, 256, 0, stream>>>(y2, bn_sc, bn_sh, x3, nullptr, 11, 11, 5, 5);
        k_conv_wmma   <<<NB, 256, 0, stream>>>(x3, w3frag, c3b, y3, 5, 5, 3, 3);
        k_stats       <<<64, 256, 0, stream>>>(y3, 9, bn3g, bn3b, bn_sc, bn_sh);
        k_bnpool      <<<(NB*64 + 255)/256, 256, 0, stream>>>(y3, bn_sc, bn_sh, nullptr, feats[s], 3, 3, 1, 1);
    }

    k_gates<<<NB, 256, 0, stream>>>(ftr, wi_f, bi_f, bh_f, xw_f, 0);
    k_gates<<<NB, 256, 0, stream>>>(ftr, wi_b, bi_b, bh_b, xw_b, 0);
    k_gates<<<NB, 256, 0, stream>>>(fte, att_wih, att_b, nullptr, xa, 1);

    size_t lstmLds = (size_t)(16384 + 64 + 64 + 256) * sizeof(float);
    k_lstm<<<1, 256, lstmLds, stream>>>(xw_f, wh_f, h0,      c0,      hf, 0);
    k_lstm<<<1, 256, lstmLds, stream>>>(xw_b, wh_b, h0 + 64, c0 + 64, hb, 1);
    k_combine<<<NB, 64, 0, stream>>>(hf, hb, ftr, ctr16, ntr);

    size_t attLds = (size_t)262144 + (size_t)(1024 + 1024 + 64*4 + 256 + 34) * sizeof(float);
    k_attlstm<<<1, 1024, attLds, stream>>>(ctr16, xa, fte, att_whh, att_h0, att_c0, att_r0, cte16);
    k_norms16<<<NB, 64, 0, stream>>>(cte16, nte);

    k_sim<<<dim3(64, 16), 128, 0, stream>>>(cte16, ctr16, nte, ntr, sim);
    k_softmax_label<<<NB, 256, 0, stream>>>(sim, trainLabel, out);
}
